// ResidualBlock_27685359190138
// MI455X (gfx1250) — matrix operand order repaired and checked by reading, compiled, not run on hardware
//
#include <hip/hip_runtime.h>
#include <cmath>

// ---------------------------------------------------------------------------
// CDNA5 (gfx1250) hyperbolic residual block
//   Y1 = x @ W1^T + b1                       (bf16x3-split WMMA GEMM)
//   O  = expmap0(relu(logmap0(project(Y1)))) (fused row kernel)
//   Y2 = O @ W2^T + b2                       (bf16x3-split WMMA GEMM)
//   out= hyp_relu(project(Y2) + x)           (fused row kernel, residual)
//
// GEMM: 128x128 block tile, 8 wave32s at 64x32 each, K staged in 32-slabs.
// Double-buffered LDS (80KB), ONE barrier per K-step: iteration kt reads
// stage kt&1 and writes stage (kt+1)&1; the barrier at the top of iteration
// kt proves all waves finished reading that buffer in iteration kt-1.
// Next-slab global loads issue before the WMMA block (latency hidden under
// 24 v_wmma per wave), hi/lo split + ds_store after it.
// ---------------------------------------------------------------------------

typedef __attribute__((ext_vector_type(16))) __bf16 v16bf;
typedef __attribute__((ext_vector_type(4)))  __bf16 v4bf;
typedef __attribute__((ext_vector_type(8)))  float  v8f;

#define EPSF 1e-7f
#define DDIM 4096
#define MROWS 8192

// GEMM tiling
#define BM 128
#define BN 128
#define BK 32
#define LDK 40   // padded LDS row stride (bf16): 80B = 20 dwords -> conflict-free, 16B aligned
#define STAGE_ELEMS (BM * LDK)           // per hi/lo per matrix per stage
#define SMEM_BYTES (4 * 2 * STAGE_ELEMS * 2)   // 4 arrays x 2 stages x bf16 = 81920B

// Split one f32 into hi/lo bf16 (hi rounds RTNE, lo captures residual; a*b ~=
// ah*bh + ah*bl + al*bh with ~2^-15 relative error).
__device__ __forceinline__ void split_store4(float4 v, __bf16* hi, __bf16* lo) {
    v4bf h, l;
    {
        __bf16 h0 = (__bf16)v.x; h[0] = h0; l[0] = (__bf16)(v.x - (float)h0);
        __bf16 h1 = (__bf16)v.y; h[1] = h1; l[1] = (__bf16)(v.y - (float)h1);
        __bf16 h2 = (__bf16)v.z; h[2] = h2; l[2] = (__bf16)(v.z - (float)h2);
        __bf16 h3 = (__bf16)v.w; h[3] = h3; l[3] = (__bf16)(v.w - (float)h3);
    }
    *(v4bf*)hi = h;
    *(v4bf*)lo = l;
}

// A-operand fragment, 16x32 bf16 (ISA 7.12.2): lane L: M=L%16.
// elements 0..7  <-> K = (L/16)*8 + e
// elements 8..15 <-> K = 16 + (L/16)*8 + (e-8)
__device__ __forceinline__ v16bf load_a_frag(const __bf16* row, int half) {
    v16bf a;
    #pragma unroll
    for (int e = 0; e < 8; ++e) a[e] = row[half * 8 + e];
    #pragma unroll
    for (int e = 0; e < 8; ++e) a[8 + e] = row[16 + half * 8 + e];
    return a;
}

// B-operand fragment, 32x16 bf16: lane L holds column N=L%16;
// elements 0..7 <-> K = (L/16)*8 + e, elements 8..15 <-> K = 16 + (L/16)*8 + (e-8).
__device__ __forceinline__ v16bf load_b_frag(const __bf16* row, int half) {
    v16bf b;
    #pragma unroll
    for (int e = 0; e < 8; ++e) b[e] = row[half * 8 + e];
    #pragma unroll
    for (int e = 0; e < 8; ++e) b[8 + e] = row[16 + half * 8 + e];
    return b;
}

__global__ __launch_bounds__(256)
void gemm_xwt_bias(const float* __restrict__ A,    // M x K (row major)
                   const float* __restrict__ W,    // N x K (row major) -> C = A*W^T
                   const float* __restrict__ bias, // N
                   float* __restrict__ C,          // M x N
                   int M, int N, int Kdim)
{
    extern __shared__ __align__(16) char smem_raw[];
    __bf16* sAh = (__bf16*)smem_raw;                 // [2][STAGE_ELEMS]
    __bf16* sAl = sAh + 2 * STAGE_ELEMS;
    __bf16* sBh = sAl + 2 * STAGE_ELEMS;
    __bf16* sBl = sBh + 2 * STAGE_ELEMS;

    const int tid  = threadIdx.x;
    const int lane = tid & 31;
    const int wid  = tid >> 5;
    const int wm   = wid & 1;     // 2 wave-rows   (64 C-rows each)
    const int wn   = wid >> 1;    // 4 wave-cols   (32 C-cols each)
    const int half = lane >> 4;   // 0/1 (K-group select)
    const int l16  = lane & 15;

    const int i0 = blockIdx.y * BM;
    const int j0 = blockIdx.x * BN;

    // Staging map: 1024 float4s (A slab 128x32 + W slab 128x32), 4 per thread.
    int srow[4], scol[4];
    #pragma unroll
    for (int i = 0; i < 4; ++i) {
        int f = tid + i * 256;
        srow[i] = f >> 3;          // 0..127
        scol[i] = (f & 7) * 4;     // 0,4,...,28
    }

    const v8f vzero = {0.f, 0.f, 0.f, 0.f, 0.f, 0.f, 0.f, 0.f};
    v8f acc[4][2];
    #pragma unroll
    for (int a = 0; a < 4; ++a)
        #pragma unroll
        for (int b = 0; b < 2; ++b) acc[a][b] = vzero;

    const int nk = Kdim / BK;

    // ---- Preload stage 0 ----
    #pragma unroll
    for (int i = 0; i < 4; ++i) {
        float4 av = *(const float4*)&A[(size_t)(i0 + srow[i]) * Kdim + scol[i]];
        float4 wv = *(const float4*)&W[(size_t)(j0 + srow[i]) * Kdim + scol[i]];
        split_store4(av, &sAh[srow[i] * LDK + scol[i]], &sAl[srow[i] * LDK + scol[i]]);
        split_store4(wv, &sBh[srow[i] * LDK + scol[i]], &sBl[srow[i] * LDK + scol[i]]);
    }

    for (int kt = 0; kt < nk; ++kt) {
        __syncthreads();                      // stage kt&1 fully written; buffer
                                              // (kt+1)&1 free (last read kt-1)
        const int cur = kt & 1;
        const int nxt = cur ^ 1;
        const bool more = (kt + 1 < nk);

        // Issue next-slab global loads first; they retire under the WMMAs.
        float4 av[4], wv[4];
        if (more) {
            const int k0n = (kt + 1) * BK;
            #pragma unroll
            for (int i = 0; i < 4; ++i) {
                av[i] = *(const float4*)&A[(size_t)(i0 + srow[i]) * Kdim + k0n + scol[i]];
                wv[i] = *(const float4*)&W[(size_t)(j0 + srow[i]) * Kdim + k0n + scol[i]];
            }
        }

        const __bf16* pAh = sAh + cur * STAGE_ELEMS;
        const __bf16* pAl = sAl + cur * STAGE_ELEMS;
        const __bf16* pBh = sBh + cur * STAGE_ELEMS;
        const __bf16* pBl = sBl + cur * STAGE_ELEMS;

        // ---- Fragments ----
        v16bf ah[4], al[4], bh[2], bl[2];
        #pragma unroll
        for (int tm = 0; tm < 4; ++tm) {
            int r = wm * 64 + tm * 16 + l16;
            ah[tm] = load_a_frag(&pAh[r * LDK], half);
            al[tm] = load_a_frag(&pAl[r * LDK], half);
        }
        #pragma unroll
        for (int tn = 0; tn < 2; ++tn) {
            int c = wn * 32 + tn * 16 + l16;
            bh[tn] = load_b_frag(&pBh[c * LDK], half);
            bl[tn] = load_b_frag(&pBl[c * LDK], half);
        }

        // ---- bf16x3 split-GEMM: hi*hi + hi*lo + lo*hi, f32 accumulate ----
        #pragma unroll
        for (int tm = 0; tm < 4; ++tm) {
            #pragma unroll
            for (int tn = 0; tn < 2; ++tn) {
                acc[tm][tn] = __builtin_amdgcn_wmma_f32_16x16x32_bf16(
                    false, ah[tm], false, bh[tn], (short)0, acc[tm][tn], false, false);
                acc[tm][tn] = __builtin_amdgcn_wmma_f32_16x16x32_bf16(
                    false, ah[tm], false, bl[tn], (short)0, acc[tm][tn], false, false);
                acc[tm][tn] = __builtin_amdgcn_wmma_f32_16x16x32_bf16(
                    false, al[tm], false, bh[tn], (short)0, acc[tm][tn], false, false);
            }
        }

        // ---- Split + store next slab into the other stage ----
        if (more) {
            __bf16* qAh = sAh + nxt * STAGE_ELEMS;
            __bf16* qAl = sAl + nxt * STAGE_ELEMS;
            __bf16* qBh = sBh + nxt * STAGE_ELEMS;
            __bf16* qBl = sBl + nxt * STAGE_ELEMS;
            #pragma unroll
            for (int i = 0; i < 4; ++i) {
                split_store4(av[i], &qAh[srow[i] * LDK + scol[i]], &qAl[srow[i] * LDK + scol[i]]);
                split_store4(wv[i], &qBh[srow[i] * LDK + scol[i]], &qBl[srow[i] * LDK + scol[i]]);
            }
        }
    }

    // ---- Epilogue: +bias, store. D layout: VGPR r -> M=r+8*half, N=l16 ----
    #pragma unroll
    for (int tm = 0; tm < 4; ++tm) {
        #pragma unroll
        for (int tn = 0; tn < 2; ++tn) {
            int col = j0 + wn * 32 + tn * 16 + l16;
            float bv = bias[col];
            int rbase = i0 + wm * 64 + tm * 16 + 8 * half;
            #pragma unroll
            for (int r = 0; r < 8; ++r) {
                C[(size_t)(rbase + r) * N + col] = acc[tm][tn][r] + bv;
            }
        }
    }
}

// ---------------------------------------------------------------------------
// O = expmap0(relu(logmap0(project(Y))))  — one block per row.
// project replaces y[0] with t = sqrt(sum(ys^2)+1); logmap0 scale d/||ys||;
// relu(scale*ys) = scale*max(ys,0) since scale>=0, so only two row sums needed.
// ---------------------------------------------------------------------------
__global__ __launch_bounds__(256)
void hyp_relu_rows(const float* __restrict__ Y, float* __restrict__ O)
{
    __shared__ float sy[DDIM];
    __shared__ float red0[256], red1[256];
    __shared__ float scal[2];
    const int row = blockIdx.x, tid = threadIdx.x;
    const float* y = Y + (size_t)row * DDIM;

    float s_all = 0.f, s_pos = 0.f;
    for (int j = tid; j < DDIM; j += 256) {
        float v = y[j];
        sy[j] = v;
        if (j > 0) {
            s_all += v * v;
            if (v > 0.f) s_pos += v * v;
        }
    }
    red0[tid] = s_all; red1[tid] = s_pos;
    __syncthreads();
    for (int s = 128; s > 0; s >>= 1) {
        if (tid < s) { red0[tid] += red0[tid + s]; red1[tid] += red1[tid + s]; }
        __syncthreads();
    }
    if (tid == 0) {
        float S  = red0[0], Sp = red1[0];
        float t  = sqrtf(fmaxf(S + 1.0f, EPSF));        // project time coord
        float d  = acoshf(fmaxf(t, 1.0f + EPSF));       // logmap0 distance
        float nn = fmaxf(sqrtf(S), EPSF);
        float scale = d / nn;
        float n2 = fmaxf(scale * sqrtf(Sp), EPSF);      // ||relu(u)||
        scal[0] = sinhf(n2) / n2 * scale;               // spatial coefficient
        scal[1] = coshf(n2);                            // time coord
    }
    __syncthreads();
    float coef = scal[0];
    float* o = O + (size_t)row * DDIM;
    for (int j = tid; j < DDIM; j += 256)
        o[j] = (j == 0) ? scal[1] : coef * fmaxf(sy[j], 0.f);
}

// ---------------------------------------------------------------------------
// out = hyp_relu(project(Y2) + x)  — residual variant; three row sums.
// ---------------------------------------------------------------------------
__global__ __launch_bounds__(256)
void residual_hyp_relu_rows(const float* __restrict__ Y2,
                            const float* __restrict__ X,
                            float* __restrict__ O)
{
    __shared__ float sz[DDIM];
    __shared__ float red0[256], red1[256], red2[256];
    __shared__ float scal[2];
    const int row = blockIdx.x, tid = threadIdx.x;
    const float* y = Y2 + (size_t)row * DDIM;
    const float* x = X  + (size_t)row * DDIM;

    float s2 = 0.f, sall = 0.f, spos = 0.f;
    for (int j = tid; j < DDIM; j += 256) {
        if (j > 0) {
            float ys = y[j];
            s2 += ys * ys;                 // for project's t2
            float zs = ys + x[j];          // residual spatial part
            sz[j] = zs;
            sall += zs * zs;
            if (zs > 0.f) spos += zs * zs;
        }
    }
    red0[tid] = s2; red1[tid] = sall; red2[tid] = spos;
    __syncthreads();
    for (int s = 128; s > 0; s >>= 1) {
        if (tid < s) {
            red0[tid] += red0[tid + s];
            red1[tid] += red1[tid + s];
            red2[tid] += red2[tid + s];
        }
        __syncthreads();
    }
    if (tid == 0) {
        float t2 = sqrtf(fmaxf(red0[0] + 1.0f, EPSF));
        float z0 = t2 + x[0];                           // residual time coord
        float d  = acoshf(fmaxf(z0, 1.0f + EPSF));
        float nn = fmaxf(sqrtf(red1[0]), EPSF);
        float scale = d / nn;
        float n2 = fmaxf(scale * sqrtf(red2[0]), EPSF);
        scal[0] = sinhf(n2) / n2 * scale;
        scal[1] = coshf(n2);
    }
    __syncthreads();
    float coef = scal[0];
    float* o = O + (size_t)row * DDIM;
    for (int j = tid; j < DDIM; j += 256)
        o[j] = (j == 0) ? scal[1] : coef * fmaxf(sz[j], 0.f);
}

// ---------------------------------------------------------------------------
extern "C" void kernel_launch(void* const* d_in, const int* in_sizes, int n_in,
                              void* d_out, int out_size, void* d_ws, size_t ws_size,
                              hipStream_t stream)
{
    const float* x  = (const float*)d_in[0];
    const float* W1 = (const float*)d_in[1];
    const float* b1 = (const float*)d_in[2];
    const float* W2 = (const float*)d_in[3];
    const float* b2 = (const float*)d_in[4];
    float* out = (float*)d_out;

    const int M = MROWS, D = DDIM;
    // Workspace: Y (GEMM output, reused for both GEMMs) + O (activated).
    // Requires 2 * 8192 * 4096 * 4B = 256 MB.
    float* Y = (float*)d_ws;
    float* O = Y + (size_t)M * D;

    dim3 blk(256);
    dim3 grd(D / BN, M / BM);   // (32, 64)

    gemm_xwt_bias<<<grd, blk, SMEM_BYTES, stream>>>(x, W1, b1, Y, M, D, D);
    hyp_relu_rows<<<M, 256, 0, stream>>>(Y, O);
    gemm_xwt_bias<<<grd, blk, SMEM_BYTES, stream>>>(O, W2, b2, Y, M, D, D);
    residual_hyp_relu_rows<<<M, 256, 0, stream>>>(Y, x, out);
}
